// TopkPointExtractor_20066087207077
// MI455X (gfx1250) — matrix operand order repaired and checked by reading, not compiled, not run on hardware
//
#include <hip/hip_runtime.h>
#include <hip/hip_bf16.h>

typedef __bf16 v16bf __attribute__((ext_vector_type(16)));
typedef float  v8f   __attribute__((ext_vector_type(8)));
typedef unsigned int u32x4 __attribute__((ext_vector_type(4)));
typedef int          i32x4 __attribute__((ext_vector_type(4)));
typedef int          i32x8 __attribute__((ext_vector_type(8)));

#define NB    16          // batch
#define CIN   512
#define HIN   134
#define WIN   134
#define HC    128
#define WC    128
#define CENC  64
#define NPIX  (HC * WC)   // 16384
#define PNB   2048
#define DEPTH_SCALE 12.0f // max(128,128)//10
#define CHSTR (HIN * WIN) // 17956 elements between channels

// LDS layout for k_encode (dynamic shared):
//  [0,        64K)           : W_enc pre-swizzled bf16 WMMA A-fragments
//  [64K,      64K+16512)     : fm tile buffer 0 (32 rows x (128+1 pad) dwords)
//  [64K+16512,64K+2*16512)   : fm tile buffer 1
#define ROWPAD_B 516                    // 128 dwords + 1 pad dword, bytes
#define TILE_B   (32 * ROWPAD_B)        // 16512
#define WF_B     (64 * 1024)
#define SMEM_B   (WF_B + 2 * TILE_B)    // 98560

// ---------------------------------------------------------------------------
// Tensor Data Mover: DMA one (32 ch x 128 px) fp32 tile into LDS.
// D# per CDNA5 ISA §8.3-8.6:
//   g0: count=1 | lds_addr | global_addr[56:0] | type=2
//   g1: data_size=4B, pad_enable, pad_interval=128dw, pad_amount=1dw,
//       tensor_dim0=128, tensor_dim1=32, tile_dim0=128, tile_dim1=32,
//       tensor_dim0_stride=17956
//   g2/g3: zero (2D tile; higher dims unused)
// ---------------------------------------------------------------------------
static __device__ inline void tdm_load_tile(unsigned ldsOff, const float* gsrc)
{
    const unsigned long ga = (unsigned long)gsrc;
    u32x4 g0 = { 1u,                                   // count=1, user descriptor
                 ldsOff,                               // lds_addr (bytes)
                 (unsigned)(ga & 0xFFFFFFFFu),
                 (unsigned)((ga >> 32) & 0x1FFFFFFu) | (2u << 30) }; // type=2
    i32x8 g1;
    g1[0] = (2 << 16)   // data_size = 4 bytes
          | (1 << 20)   // pad_enable
          | (6 << 22)   // pad_interval: 128 dwords
          | (0 << 25);  // pad_amount: 1 dword
    g1[1] = (int)(128u << 16);   // tensor_dim0 low16  (bits 79:48)
    g1[2] = (int)(32u << 16);    // tensor_dim0 hi16=0, tensor_dim1 low16=32
    g1[3] = (int)(128u << 16);   // tensor_dim1 hi16=0, tile_dim0=128
    g1[4] = 32;                  // tile_dim1=32, tile_dim2=0
    g1[5] = CHSTR;               // tensor_dim0_stride low32
    g1[6] = 0;                   // stride hi16, dim1_stride low16
    g1[7] = 0;
    i32x4 gz = { 0, 0, 0, 0 };
#if __clang_major__ >= 23
    i32x8 gz8 = { 0, 0, 0, 0, 0, 0, 0, 0 };
    __builtin_amdgcn_tensor_load_to_lds(g0, g1, gz, gz, gz8, 0);
#else
    __builtin_amdgcn_tensor_load_to_lds(g0, g1, gz, gz, 0);
#endif
}

// ---------------------------------------------------------------------------
// Kernel 1: fused 1x1-conv GEMM (WMMA bf16 -> f32 accum), sum-of-squares,
// depth dot.  One WG per (batch,row): 256 threads = 8 waves; wave w owns
// pixels [16w,16w+16).  B-tiles arrive via TDM double buffering; A-fragments
// are staged once in LDS pre-swizzled to the ISA 16-bit A layout.
// ---------------------------------------------------------------------------
__global__ __launch_bounds__(256)
void k_encode(const float* __restrict__ fm, const float* __restrict__ Wenc,
              const float* __restrict__ benc, const float* __restrict__ Wd,
              const float* __restrict__ bd,
              float* __restrict__ pf2,   // (B, NPIX, 64)
              float* __restrict__ xbuf,  // (B, NPIX)
              float* __restrict__ dbuf)  // (B, NPIX)
{
    extern __shared__ char smem[];
    char* Wf    = smem;            // A fragments (bf16)
    char* tiles = smem + WF_B;     // two fm tile buffers

    const int wg   = blockIdx.x;
    const int b    = wg >> 7;
    const int h    = wg & (HC - 1);
    const int tid  = threadIdx.x;
    const int wave = tid >> 5;
    const int lane = tid & 31;
    const int n    = lane & 15;
    const int hi   = lane >> 4;
    const int pixcol = wave * 16 + n;

    // ---- Stage W_enc once: pre-swizzled into WMMA A-fragment order.
    // Pair index p = s*1024 + mt*256 + l*8 + e2 (e = 2*e2); fragment element
    // (s,mt,lane,e) maps to (m = mt*16 + (l&15), k = s*32 + 16*(e>>3) + (l>>4)*8 + (e&7)).
    {
        unsigned* W32 = (unsigned*)Wf;
        for (int p = tid; p < 16384; p += 256) {
            const int e2 = p & 7, l = (p >> 3) & 31, mt = (p >> 8) & 3, s = p >> 10;
            const int e  = 2 * e2;
            const int m  = mt * 16 + (l & 15);
            const int k  = s * 32 + ((e >> 3) << 4) + (l >> 4) * 8 + (e & 7);
            union { unsigned u; __bf16 h2[2]; } cv;
            cv.h2[0] = (__bf16)Wenc[m * CIN + k];
            cv.h2[1] = (__bf16)Wenc[m * CIN + k + 1];
            W32[p] = cv.u;
        }
    }

    const float* fmRow = fm + (long)b * CIN * CHSTR + (long)(h + 3) * WIN + 3;

    // Prime the TDM pipeline: tile for k-step 0 into buffer 0.
    if (wave == 0) tdm_load_tile(WF_B, fmRow);
    __syncthreads();   // also publishes the staged A-fragments

    v8f acc[4];
    #pragma unroll
    for (int mt = 0; mt < 4; ++mt)
        acc[mt] = (v8f){0.f,0.f,0.f,0.f,0.f,0.f,0.f,0.f};
    float ddot = 0.f;

    for (int s = 0; s < 16; ++s) {          // K in steps of 32 (CIN=512)
        const int kb = s * 32;
        char* cur = tiles + (s & 1) * TILE_B;

        if (wave == 0) {
            if (s + 1 < 16) {               // prefetch next tile, keep 1 in flight
                tdm_load_tile(WF_B + ((s + 1) & 1) * TILE_B,
                              fmRow + (long)(kb + 32) * CHSTR);
                __builtin_amdgcn_s_wait_tensorcnt(1);
            } else {
                __builtin_amdgcn_s_wait_tensorcnt(0);
            }
        }
        __syncthreads();                    // tile `cur` visible to all waves

        // B fragment (32x16 bf16) from padded LDS tile; feed depth dot too.
        v16bf bm;
        #pragma unroll
        for (int e = 0; e < 16; ++e) {
            const int   k = ((e >> 3) << 4) + hi * 8 + (e & 7);
            const float f = *(const float*)(cur + k * ROWPAD_B + pixcol * 4);
            ddot += f * Wd[kb + k];         // each (k,pix) consumed exactly once
            bm[e] = (__bf16)f;
        }
        // A fragments: one aligned 32B LDS read each.
        #pragma unroll
        for (int mt = 0; mt < 4; ++mt) {
            const v16bf am =
                *(const v16bf*)(Wf + (((s * 4 + mt) * 32 + lane) << 5));
            acc[mt] = __builtin_amdgcn_wmma_f32_16x16x32_bf16(
                false, am, false, bm, (short)0, acc[mt], false, false);
        }
        __syncthreads();                    // retire `cur` before TDM reuses it
    }

    // Epilogue: bias, store pf (point-major), sum of squares, depth.
    const int  pixIdx = h * WC + pixcol;
    const long rowOff = ((long)(b * NPIX + pixIdx)) * CENC;
    float sq = 0.f;
    #pragma unroll
    for (int mt = 0; mt < 4; ++mt) {
        #pragma unroll
        for (int r = 0; r < 8; ++r) {
            const int   m = mt * 16 + hi * 8 + r;   // C/D layout: M = r + 8*hi
            const float v = acc[mt][r] + benc[m];
            pf2[rowOff + m] = v;
            sq += v * v;
        }
    }
    sq   += __shfl_xor(sq,   16, 32);
    ddot += __shfl_xor(ddot, 16, 32);
    if (hi == 0) {
        xbuf[b * NPIX + pixIdx] = sq;
        dbuf[b * NPIX + pixIdx] = tanhf(ddot + bd[0]) * DEPTH_SCALE;
    }
}

// ---------------------------------------------------------------------------
// Kernel 2: per-batch top-2048 of 16384 non-negative floats.
// Radix-select (4 x 8-bit MSB-first) for the k-th key, compact survivors,
// bitonic-sort 2048 packed (value desc, index asc) keys in LDS.
// ---------------------------------------------------------------------------
__global__ __launch_bounds__(1024)
void k_topk(const float* __restrict__ xbuf, int* __restrict__ inds)
{
    const int b   = blockIdx.x;
    const int tid = threadIdx.x;
    const float* x = xbuf + b * NPIX;

    __shared__ unsigned hist[256];
    __shared__ unsigned s_prefix, s_want, s_cntG, s_pG, s_pT;
    __shared__ unsigned long long sel[PNB];   // 16 KB

    if (tid == 0) { s_prefix = 0u; s_want = PNB; }
    __syncthreads();

    for (int shift = 24; shift >= 0; shift -= 8) {
        if (tid < 256) hist[tid] = 0u;
        __syncthreads();
        const unsigned pfx = s_prefix;
        for (int i = tid; i < NPIX; i += 1024) {
            const unsigned key  = __float_as_uint(x[i]); // keys >= 0
            const bool     live = (shift == 24) ||
                                  (((key ^ pfx) >> (shift + 8)) == 0u);
            if (live) atomicAdd(&hist[(key >> shift) & 255u], 1u);
        }
        __syncthreads();
        if (tid == 0) {
            unsigned cum = 0; int d = 255;
            for (; d > 0; --d) {
                const unsigned c = hist[d];
                if (cum + c >= s_want) break;
                cum += c;
            }
            s_prefix |= ((unsigned)d) << shift;
            s_want   -= cum;
        }
        __syncthreads();
    }

    const unsigned T = s_prefix;  // key of the 2048-th largest element
    if (tid == 0) { s_cntG = 0u; s_pG = 0u; s_pT = 0u; }
    __syncthreads();
    for (int i = tid; i < NPIX; i += 1024)
        if (__float_as_uint(x[i]) > T) atomicAdd(&s_cntG, 1u);
    __syncthreads();
    const unsigned G = s_cntG;
    __syncthreads();

    for (int i = tid; i < NPIX; i += 1024) {
        const unsigned key = __float_as_uint(x[i]);
        const unsigned long long pack =
            ((unsigned long long)key << 32) | (unsigned)(0xFFFFFFFFu - (unsigned)i);
        if (key > T) {
            sel[atomicAdd(&s_pG, 1u)] = pack;
        } else if (key == T) {
            const unsigned p = atomicAdd(&s_pT, 1u);
            if (G + p < PNB) sel[G + p] = pack;
        }
    }
    __syncthreads();

    for (unsigned k = 2; k <= PNB; k <<= 1) {
        for (unsigned j = k >> 1; j > 0; j >>= 1) {
            const unsigned i1 = (tid / j) * (j * 2) + (tid % j);
            const unsigned i2 = i1 + j;
            const bool desc = ((i1 & k) == 0);
            const unsigned long long a = sel[i1], c = sel[i2];
            const bool sw = desc ? (a < c) : (a > c);
            if (sw) { sel[i1] = c; sel[i2] = a; }
            __syncthreads();
        }
    }

    for (int t = tid; t < PNB; t += 1024)
        inds[b * PNB + t] =
            (int)(0xFFFFFFFFu - (unsigned)(sel[t] & 0xFFFFFFFFull));
}

// ---------------------------------------------------------------------------
// Kernel 3: gather output rows: [abs, ord, depth, 64 features] = 67 floats.
// ---------------------------------------------------------------------------
__global__ __launch_bounds__(64)
void k_gather(const float* __restrict__ pf2, const float* __restrict__ dbuf,
              const int* __restrict__ inds, float* __restrict__ out)
{
    const int m = blockIdx.x;       // b*PNB + j
    const int b = m >> 11;
    const int t = threadIdx.x;
    const int idx = inds[m];
    float* o = out + (long)m * (CENC + 3);
    o[3 + t] = pf2[((long)(b * NPIX + idx)) * CENC + t];
    if (t == 0) {
        o[0] = (float)(idx % WC);
        o[1] = (float)(idx / WC);
        o[2] = dbuf[b * NPIX + idx];
    }
}

// ---------------------------------------------------------------------------
extern "C" void kernel_launch(void* const* d_in, const int* in_sizes, int n_in,
                              void* d_out, int out_size, void* d_ws, size_t ws_size,
                              hipStream_t stream)
{
    (void)in_sizes; (void)n_in; (void)out_size; (void)ws_size;
    const float* fm   = (const float*)d_in[0];
    const float* Wenc = (const float*)d_in[1];
    const float* benc = (const float*)d_in[2];
    const float* Wd   = (const float*)d_in[3];
    const float* bd   = (const float*)d_in[4];
    float* out = (float*)d_out;

    char*  ws   = (char*)d_ws;
    float* pf2  = (float*)ws;                                   // 64 MB
    float* xbuf = (float*)(ws + (size_t)NB * NPIX * CENC * 4);  // 1 MB
    float* dbuf = xbuf + (size_t)NB * NPIX;                     // 1 MB
    int*   inds = (int*)(dbuf + (size_t)NB * NPIX);             // 128 KB

    k_encode<<<NB * HC, 256, SMEM_B, stream>>>(fm, Wenc, benc, Wd, bd,
                                               pf2, xbuf, dbuf);
    k_topk  <<<NB, 1024, 0, stream>>>(xbuf, inds);
    k_gather<<<NB * PNB, 64, 0, stream>>>(pf2, dbuf, inds, out);
}
